// Aggregator_55954833932569
// MI455X (gfx1250) — compile-verified, matrix operand order checked by reading, not run on hardware
//
#include <hip/hip_runtime.h>

typedef __attribute__((ext_vector_type(16))) _Float16 v16h;
typedef __attribute__((ext_vector_type(8)))  float    v8f;

#define FEAT 128
#define KN 32
#define BATCH_N 50000
#define WAVES_PER_BLOCK 8

// One wave32 per batch row.
// out[row, :] = (features[node] + sum_k features[neigh[row,k]]) / 33
// Neighbor sum done with V_WMMA_F32_16X16X32_F16:
//   A[16x32] = f16(features) for 16 feature columns x 32 neighbors
//   B[32x16] = all ones (rank-1 -> all D columns identical; layout-agnostic)
//   C/D f32 accumulate -> only input-rounding error, exact sum in f32.
__global__ __launch_bounds__(256) void agg_wmma_kernel(
    const float* __restrict__ features,
    const int*   __restrict__ nodes,
    const int*   __restrict__ neigh,
    float*       __restrict__ out)
{
  const int lane = threadIdx.x & 31;
  const int wave = threadIdx.x >> 5;
  const int row  = blockIdx.x * WAVES_PER_BLOCK + wave;
  if (row >= BATCH_N) return;                 // wave-uniform guard

  const int g = lane >> 4;                    // half-wave group
  const int m = lane & 15;                    // feature column within 16-block

  // The 16 K-slots this lane owns in the 16-bit A[16x32] layout:
  //   VGPR v in 0..3 -> K = 8g + 2v, 8g + 2v + 1
  //   VGPR v in 4..7 -> K = 16 + 8g + 2(v-4), +1
  long base[16];
  #pragma unroll
  for (int v = 0; v < 8; ++v) {
    const int kv = (v < 4) ? (8 * g + 2 * v) : (16 + 8 * g + 2 * (v - 4));
    const int n0 = neigh[row * KN + kv];
    const int n1 = neigh[row * KN + kv + 1];
    base[2 * v]     = (long)n0 * FEAT;
    base[2 * v + 1] = (long)n1 * FEAT;
  }

  v16h bones;
  #pragma unroll
  for (int i = 0; i < 16; ++i) bones[i] = (_Float16)1.0f;

  const int  node     = nodes[row];
  const long selfBase = (long)node * FEAT;
  const long outBase  = (long)row * FEAT;
  const float inv33   = 1.0f / 33.0f;

  #pragma unroll 2
  for (int blk = 0; blk < 8; ++blk) {
    const int col = blk * 16 + m;

    // Build A fragment: each load is two coalesced 64B segments
    // (lanes 0-15 one neighbor row, lanes 16-31 another).
    v16h a;
    #pragma unroll
    for (int t = 0; t < 16; ++t)
      a[t] = (_Float16)features[base[t] + col];

    v8f c = {};
    c = __builtin_amdgcn_wmma_f32_16x16x32_f16(
        /*neg_a=*/false, a, /*neg_b=*/false, bones,
        /*c_mod=*/(short)0, c, /*reuse_a=*/false, /*reuse_b=*/false);

    // D[m,n] identical over n. Lane 0: c[j] = D[j,0] (features blk*16+0..7),
    // lane 16: c[j] = D[8+j,0] (features blk*16+8..15).
    if ((lane & 15) == 0) {
      const int fbase = blk * 16 + g * 8;
      #pragma unroll
      for (int j = 0; j < 8; ++j) {
        const float s = c[j] + features[selfBase + fbase + j];
        out[outBase + fbase + j] = s * inv33;
      }
    }
  }
}

extern "C" void kernel_launch(void* const* d_in, const int* in_sizes, int n_in,
                              void* d_out, int out_size, void* d_ws, size_t ws_size,
                              hipStream_t stream) {
  const float* features = (const float*)d_in[0];   // [1M, 128] f32
  const int*   nodes    = (const int*)d_in[1];     // [50000] i32
  const int*   neigh    = (const int*)d_in[2];     // [50000, 32] i32
  float*       out      = (float*)d_out;           // [50000, 128] f32

  const int blocks = (BATCH_N + WAVES_PER_BLOCK - 1) / WAVES_PER_BLOCK; // 6250
  agg_wmma_kernel<<<blocks, 32 * WAVES_PER_BLOCK, 0, stream>>>(
      features, nodes, neigh, out);
}
